// IEGMNetXNORNew_283467841702
// MI455X (gfx1250) — matrix operand order repaired and checked by reading, not compiled, not run on hardware
//
#include <hip/hip_runtime.h>

typedef __attribute__((ext_vector_type(16))) _Float16 v16h;
typedef __attribute__((ext_vector_type(8)))  _Float16 v8h;
typedef __attribute__((ext_vector_type(8)))  float    v8f;

// ---------------- Problem constants ----------------
#define NB 8192           // batch
#define L0 1250
#define L1 623            // (1250-6)/2+1
#define L2 310            // (623-5)/2+1
#define L3 154            // (310-4)/2+1
#define L4 76             // (154-4)/2+1
#define L5 37             // (76-4)/2+1
#define KPAD 768          // 740 padded to 24*32 for WMMA K-loop

// workspace layout (bytes)
#define OFF_STATS 0u            // 7*64 f32   (per-layer sum/sumsq pairs)
#define OFF_SS    4096u         // 7*64 f32   (per-layer scale/shift)
#define OFF_W6    8192u         // 24*32*16 f16 = 24576 B (FC1 B-fragments)
#define OFF_Y6    40960u        // 8192*10 f32
#define OFF_Y7    368640u       // 8192*2  f32
#define OFF_BUFA  (1u<<20)      // up to 8192*3*623 f32 = 61.3 MB
#define OFF_BUFB  (64u<<20)     // up to 8192*5*310 f32 = 50.8 MB

// 4-bit uniform quantizer, eval mode. round-to-nearest-even matches jnp.round.
// Also absorbs htanh: clip(-1,1) then clip(-1,0.875) == clip(-1,0.875).
__device__ __forceinline__ float quant4(float x) {
    float xc = fminf(fmaxf(x, -1.0f), 0.875f);
    return rintf(xc * 8.0f) * 0.125f;
}

// ---------------- utility: zero stats ----------------
__global__ void zero_kernel(float* p, int n) {
    int i = blockIdx.x * blockDim.x + threadIdx.x;
    if (i < n) p[i] = 0.0f;
}

// ---------------- fused BN-affine + quantize + direct conv + stat reduce ---
// in : raw pre-BN output of previous layer [B][CIN][LIN] (or x for FIRST)
// out: raw pre-BN output of this layer     [B][COUT][LOUT]
// ss : scale[CIN], shift[CIN] of previous layer's BN (unused if FIRST)
// stats: 2*COUT accumulators (sum, sumsq) for this layer's BN
template <int CIN, int COUT, int KH, int STRIDE, int LIN, int LOUT, bool FIRST>
__global__ void conv_bn_quant_kernel(const float* __restrict__ in,
                                     const float* __restrict__ w,
                                     const float* __restrict__ ss,
                                     float* __restrict__ out,
                                     float* __restrict__ stats) {
    const int o   = blockIdx.y;
    const int tid = threadIdx.x;

    __shared__ float qw[CIN * KH];
    __shared__ float ssc[CIN];
    __shared__ float ssh[CIN];
    for (int i = tid; i < CIN * KH; i += blockDim.x)
        qw[i] = quant4(w[o * CIN * KH + i]);
    if (!FIRST) {
        for (int i = tid; i < CIN; i += blockDim.x) {
            ssc[i] = ss[i];
            ssh[i] = ss[CIN + i];
        }
    }
    __syncthreads();

    const int idx   = blockIdx.x * blockDim.x + tid;
    const bool valid = idx < NB * LOUT;
    float y = 0.0f;
    if (valid) {
        const int b = idx / LOUT;
        const int t = idx % LOUT;
        const float* xin = in + (size_t)b * CIN * LIN;
        float acc = 0.0f;
#pragma unroll
        for (int ci = 0; ci < CIN; ++ci) {
            const float* xr = xin + ci * LIN + t * STRIDE;
#pragma unroll
            for (int k = 0; k < KH; ++k) {
                float xv = xr[k];
                float a  = FIRST ? quant4(xv)
                                 : quant4(fmaf(xv, ssc[ci], ssh[ci]));
                acc = fmaf(a, qw[ci * KH + k], acc);
            }
        }
        out[(size_t)b * COUT * LOUT + (size_t)o * LOUT + t] = acc;
        y = acc;
    }

    // block reduction of sum / sumsq for this out-channel
    __shared__ float r0[256];
    __shared__ float r1[256];
    r0[tid] = valid ? y : 0.0f;
    r1[tid] = valid ? y * y : 0.0f;
    __syncthreads();
    for (int s = 128; s > 0; s >>= 1) {
        if (tid < s) { r0[tid] += r0[tid + s]; r1[tid] += r1[tid + s]; }
        __syncthreads();
    }
    if (tid == 0) {
        atomicAdd(&stats[2 * o + 0], r0[0]);
        atomicAdd(&stats[2 * o + 1], r1[0]);
    }
}

// ---------------- BN finalize: stats -> (scale, shift) ----------------
template <int C>
__global__ void bn_finalize_kernel(const float* __restrict__ stats,
                                   const float* __restrict__ g,
                                   const float* __restrict__ b,
                                   float invN, float* __restrict__ ss) {
    int c = threadIdx.x;
    if (c < C) {
        float mean = stats[2 * c + 0] * invN;
        float var  = stats[2 * c + 1] * invN - mean * mean;  // biased, matches jnp.var
        float sc   = g[c] * rsqrtf(var + 1e-5f);
        ss[c]      = sc;
        ss[C + c]  = b[c] - mean * sc;
    }
}

// ---------------- pack FC1 activations: bn5-affine + quantize -> f16 -------
// a6[b][k], k = c*37 + t (flatten order of NCHW reshape), zero pad k in [740,768)
__global__ void pack_a6_kernel(const float* __restrict__ y5,
                               const float* __restrict__ ss5,
                               _Float16* __restrict__ a6) {
    int idx = blockIdx.x * blockDim.x + threadIdx.x;   // over NB*KPAD
    if (idx >= NB * KPAD) return;
    int b = idx / KPAD, k = idx % KPAD;
    float v = 0.0f;
    if (k < 740) {
        int c = k / 37;
        float x = y5[(size_t)b * 740 + k];
        v = quant4(fmaf(x, ss5[c], ss5[20 + c]));
    }
    a6[idx] = (_Float16)v;
}

// ---------------- pack FC1 weights into per-lane WMMA B-fragments ----------
// B matrix (32x16 f16, per ISA 7.12.2 / sparse-B analogue): lane = N (mod 16),
// lanes 0-15 hold K=kc*32+{0..7,16..23}, lanes 16-31 hold K=kc*32+{8..15,24..31}, 16 halves/lane.
// w6frag index = (kc*32 + lane)*16 + j.
__global__ void pack_w6_kernel(const float* __restrict__ fw1,
                               _Float16* __restrict__ w6frag) {
    int idx = blockIdx.x * blockDim.x + threadIdx.x;   // 24*32*16 = 12288
    if (idx >= 24 * 32 * 16) return;
    int kc = idx >> 9;
    int lane = (idx >> 4) & 31;
    int j = idx & 15;
    int n = lane & 15;
    int k = kc * 32 + ((lane < 16) ? 0 : 8) + ((j < 8) ? j : j + 8);
    float v = 0.0f;
    if (n < 10 && k < 740) v = quant4(fw1[n * 740 + k]);
    w6frag[idx] = (_Float16)v;
}

// ---------------- FC1 GEMM: (8192x768) x (768x16) via WMMA f16 -------------
// One wave per 16-row tile of M; 24 accumulating v_wmma_f32_16x16x32_f16.
// A-fragment layout (ISA 7.12.2, 16-bit A 16x32): lanes 0-15 row M=lane,
// halves = K{0..7},{16..23}; lanes 16-31 row M=lane-16, K{8..15},{24..31}.
__global__ void fc1_wmma_kernel(const _Float16* __restrict__ a6,
                                const _Float16* __restrict__ w6frag,
                                float* __restrict__ y6) {
    const int wave = (blockIdx.x * blockDim.x + threadIdx.x) >> 5;
    const int lane = threadIdx.x & 31;
    const int m0 = wave * 16;
    if (m0 >= NB) return;                 // wave-uniform; EXEC stays all-ones

    const int row = m0 + (lane & 15);
    const _Float16* abase = a6 + (size_t)row * KPAD + ((lane < 16) ? 0 : 8);
    const _Float16* bbase = w6frag + (size_t)lane * 16;

    v8f c = {};
#pragma unroll
    for (int kc = 0; kc < 24; ++kc) {
        v8h alo = *(const v8h*)(abase + kc * 32);
        v8h ahi = *(const v8h*)(abase + kc * 32 + 16);
        v16h a = __builtin_shufflevector(alo, ahi, 0, 1, 2, 3, 4, 5, 6, 7,
                                         8, 9, 10, 11, 12, 13, 14, 15);
        v16h bf = *(const v16h*)(bbase + (size_t)kc * 32 * 16);
        c = __builtin_amdgcn_wmma_f32_16x16x32_f16(
                false, a, false, bf, (short)0, c, false, false);
    }

    // D layout: VGPR r -> (M = r + (lane<16?0:8), N = lane&15)
    const int n = lane & 15;
    const int rbase = m0 + ((lane < 16) ? 0 : 8);
    if (n < 10) {
#pragma unroll
        for (int r = 0; r < 8; ++r)
            y6[(size_t)(rbase + r) * 10 + n] = c[r];
    }
}

// ---------------- per-column stats over (rows x C) ----------------
template <int C>
__global__ void colstats_kernel(const float* __restrict__ y, int rows,
                                float* __restrict__ stats) {
    const int c = blockIdx.y;
    const int tid = threadIdx.x;
    float s = 0.0f, s2 = 0.0f;
    for (int r = blockIdx.x * blockDim.x + tid; r < rows;
         r += gridDim.x * blockDim.x) {
        float v = y[(size_t)r * C + c];
        s += v; s2 += v * v;
    }
    __shared__ float r0[256];
    __shared__ float r1[256];
    r0[tid] = s; r1[tid] = s2;
    __syncthreads();
    for (int st = 128; st > 0; st >>= 1) {
        if (tid < st) { r0[tid] += r0[tid + st]; r1[tid] += r1[tid + st]; }
        __syncthreads();
    }
    if (tid == 0) {
        atomicAdd(&stats[2 * c + 0], r0[0]);
        atomicAdd(&stats[2 * c + 1], r1[0]);
    }
}

// ---------------- FC2: bn6-affine + quantize, K=10, N=2, + stats ----------
__global__ void fc2_kernel(const float* __restrict__ y6,
                           const float* __restrict__ ss6,
                           const float* __restrict__ fw2,
                           float* __restrict__ y7,
                           float* __restrict__ stats7) {
    const int tid = threadIdx.x;
    const int b = blockIdx.x * blockDim.x + tid;
    float a0 = 0.0f, a1 = 0.0f;
    if (b < NB) {
#pragma unroll
        for (int k = 0; k < 10; ++k) {
            float a = quant4(fmaf(y6[(size_t)b * 10 + k], ss6[k], ss6[10 + k]));
            a0 = fmaf(a, quant4(fw2[k]), a0);
            a1 = fmaf(a, quant4(fw2[10 + k]), a1);
        }
        y7[(size_t)b * 2 + 0] = a0;
        y7[(size_t)b * 2 + 1] = a1;
    }
    __shared__ float r[4][256];
    r[0][tid] = a0; r[1][tid] = a0 * a0;
    r[2][tid] = a1; r[3][tid] = a1 * a1;
    __syncthreads();
    for (int s = 128; s > 0; s >>= 1) {
        if (tid < s)
            for (int q = 0; q < 4; ++q) r[q][tid] += r[q][tid + s];
        __syncthreads();
    }
    if (tid == 0) {
        atomicAdd(&stats7[0], r[0][0]);
        atomicAdd(&stats7[1], r[1][0]);
        atomicAdd(&stats7[2], r[2][0]);
        atomicAdd(&stats7[3], r[3][0]);
    }
}

// ---------------- final BN7 -> output ----------------
__global__ void final_out_kernel(const float* __restrict__ y7,
                                 const float* __restrict__ ss7,
                                 float* __restrict__ out) {
    int i = blockIdx.x * blockDim.x + threadIdx.x;
    if (i < NB * 2) {
        int c = i & 1;
        out[i] = fmaf(y7[i], ss7[c], ss7[2 + c]);
    }
}

extern "C" void kernel_launch(void* const* d_in, const int* in_sizes, int n_in,
                              void* d_out, int out_size, void* d_ws, size_t ws_size,
                              hipStream_t stream) {
    const float* x   = (const float*)d_in[0];
    const float* w1  = (const float*)d_in[1];
    const float* w2  = (const float*)d_in[2];
    const float* w3  = (const float*)d_in[3];
    const float* w4  = (const float*)d_in[4];
    const float* w5  = (const float*)d_in[5];
    const float* fw1 = (const float*)d_in[6];
    const float* fw2 = (const float*)d_in[7];
    const float* g[7];
    const float* bb[7];
    for (int i = 0; i < 7; ++i) {
        g[i]  = (const float*)d_in[8 + i];
        bb[i] = (const float*)d_in[15 + i];
    }

    char* ws = (char*)d_ws;
    float*    stats  = (float*)(ws + OFF_STATS);     // layer i at stats + i*64
    float*    ss     = (float*)(ws + OFF_SS);        // layer i at ss + i*64
    _Float16* w6frag = (_Float16*)(ws + OFF_W6);
    float*    y6     = (float*)(ws + OFF_Y6);
    float*    y7     = (float*)(ws + OFF_Y7);
    float*    bufA   = (float*)(ws + OFF_BUFA);      // y1, y3, y5
    float*    bufB   = (float*)(ws + OFF_BUFB);      // y2, y4, a6
    _Float16* a6     = (_Float16*)bufB;

    float* st1 = stats + 0 * 64; float* ss1 = ss + 0 * 64;
    float* st2 = stats + 1 * 64; float* ss2 = ss + 1 * 64;
    float* st3 = stats + 2 * 64; float* ss3 = ss + 2 * 64;
    float* st4 = stats + 3 * 64; float* ss4 = ss + 3 * 64;
    float* st5 = stats + 4 * 64; float* ss5 = ss + 4 * 64;
    float* st6 = stats + 5 * 64; float* ss6 = ss + 5 * 64;
    float* st7 = stats + 6 * 64; float* ss7 = ss + 6 * 64;

    zero_kernel<<<2, 256, 0, stream>>>(stats, 7 * 64);

    // conv1: (B,1,1250) -> (B,3,623)
    conv_bn_quant_kernel<1, 3, 6, 2, L0, L1, true>
        <<<dim3((NB * L1 + 255) / 256, 3), 256, 0, stream>>>(x, w1, ss1, bufA, st1);
    bn_finalize_kernel<3><<<1, 32, 0, stream>>>(st1, g[0], bb[0], 1.0f / ((float)NB * L1), ss1);

    // conv2: (B,3,623) -> (B,5,310)
    conv_bn_quant_kernel<3, 5, 5, 2, L1, L2, false>
        <<<dim3((NB * L2 + 255) / 256, 5), 256, 0, stream>>>(bufA, w2, ss1, bufB, st2);
    bn_finalize_kernel<5><<<1, 32, 0, stream>>>(st2, g[1], bb[1], 1.0f / ((float)NB * L2), ss2);

    // conv3: (B,5,310) -> (B,10,154)
    conv_bn_quant_kernel<5, 10, 4, 2, L2, L3, false>
        <<<dim3((NB * L3 + 255) / 256, 10), 256, 0, stream>>>(bufB, w3, ss2, bufA, st3);
    bn_finalize_kernel<10><<<1, 32, 0, stream>>>(st3, g[2], bb[2], 1.0f / ((float)NB * L3), ss3);

    // conv4: (B,10,154) -> (B,20,76)
    conv_bn_quant_kernel<10, 20, 4, 2, L3, L4, false>
        <<<dim3((NB * L4 + 255) / 256, 20), 256, 0, stream>>>(bufA, w4, ss3, bufB, st4);
    bn_finalize_kernel<20><<<1, 32, 0, stream>>>(st4, g[3], bb[3], 1.0f / ((float)NB * L4), ss4);

    // conv5: (B,20,76) -> (B,20,37)
    conv_bn_quant_kernel<20, 20, 4, 2, L4, L5, false>
        <<<dim3((NB * L5 + 255) / 256, 20), 256, 0, stream>>>(bufB, w5, ss4, bufA, st5);
    bn_finalize_kernel<20><<<1, 32, 0, stream>>>(st5, g[4], bb[4], 1.0f / ((float)NB * L5), ss5);

    // FC1 via WMMA
    pack_a6_kernel<<<(NB * KPAD + 255) / 256, 256, 0, stream>>>(bufA, ss5, a6);
    pack_w6_kernel<<<48, 256, 0, stream>>>(fw1, w6frag);
    fc1_wmma_kernel<<<(NB / 16) * 32 / 128, 128, 0, stream>>>(a6, w6frag, y6);

    colstats_kernel<10><<<dim3(32, 10), 256, 0, stream>>>(y6, NB, st6);
    bn_finalize_kernel<10><<<1, 32, 0, stream>>>(st6, g[5], bb[5], 1.0f / (float)NB, ss6);

    // FC2 + stats
    fc2_kernel<<<NB / 256, 256, 0, stream>>>(y6, ss6, fw2, y7, st7);
    bn_finalize_kernel<2><<<1, 32, 0, stream>>>(st7, g[6], bb[6], 1.0f / (float)NB, ss7);

    final_out_kernel<<<(NB * 2 + 255) / 256, 256, 0, stream>>>(y7, ss7, (float*)d_out);
}
